// DyCls_34488587387192
// MI455X (gfx1250) — matrix operand order repaired and checked by reading, not compiled, not run on hardware
//
#include <hip/hip_runtime.h>
#include <hip/hip_bf16.h>

// ---------------------------------------------------------------------------
// out = (softmax(mod,axis=1)+1) * x  @  W^T      B=8192, D=1024, C=4096
//
// Compute-bound (68.7 GFLOP vs ~6.5us of HBM traffic at 23.3 TB/s), so use
// the high-rate matrix path: v_wmma_f32_16x16x32_bf16 (K=32/op, f32 accum).
// bf16 A (16MB) + bf16 W (8MB) fit in the 192MB L2 -> register-blocked GEMM
// streaming fragments from WGP$/L2 directly.
//   - wave tile 64x128 (4 M-frags x 8 N-frags): 384 B of fragment traffic per
//     WMMA (vs 512 B at 64x64), A reused 8x / B reused 4x in registers.
//   - out (128MB, write-once) stored non-temporal so it does not rinse the
//     L2 working set; read-once f32 inputs loaded non-temporal.
// ---------------------------------------------------------------------------

typedef __attribute__((ext_vector_type(16))) __bf16 v16bf;
typedef __attribute__((ext_vector_type(4)))  __bf16 v4bf;
typedef __attribute__((ext_vector_type(8)))  float  v8f;
typedef __attribute__((ext_vector_type(4)))  float  v4f;

#define B_ROWS 8192
#define D_DIM  1024
#define C_DIM  4096

union Frag {
    v16bf v;
    uint4 q[2];   // two 16-byte halves
};

// ---------------------------------------------------------------------------
// Kernel 0: W (C x D f32, row-major) -> bf16
// ---------------------------------------------------------------------------
__global__ __launch_bounds__(256) void wcvt_kernel(const float* __restrict__ W,
                                                   __bf16* __restrict__ Wbf) {
    const size_t i = ((size_t)blockIdx.x * 256 + threadIdx.x) * 4;
    const v4f f = __builtin_nontemporal_load((const v4f*)(W + i));
    v4bf o;
    o[0] = (__bf16)f[0]; o[1] = (__bf16)f[1]; o[2] = (__bf16)f[2]; o[3] = (__bf16)f[3];
    *(v4bf*)(Wbf + i) = o;   // re-read by GEMM: keep regular-temporal
}

// ---------------------------------------------------------------------------
// Kernel 1: per-row softmax(mod)+1, times x, emit bf16 A (B x D row-major).
// One 256-thread block (8 wave32) per row; 4 elements/thread.
// ---------------------------------------------------------------------------
__global__ __launch_bounds__(256) void softmax_scale_kernel(
    const float* __restrict__ x, const float* __restrict__ mod,
    __bf16* __restrict__ Abf) {
    const int row  = blockIdx.x;
    const int tid  = threadIdx.x;
    const int lane = tid & 31;
    const int wave = tid >> 5;

    const float* mrow = mod + (size_t)row * D_DIM;
    const float* xrow = x   + (size_t)row * D_DIM;

    float v[4];
#pragma unroll
    for (int t = 0; t < 4; ++t)
        v[t] = __builtin_nontemporal_load(mrow + tid + t * 256);

    // ---- block max ----
    float m = fmaxf(fmaxf(v[0], v[1]), fmaxf(v[2], v[3]));
#pragma unroll
    for (int off = 16; off > 0; off >>= 1)
        m = fmaxf(m, __shfl_xor(m, off, 32));
    __shared__ float redmax[8];
    __shared__ float redsum[8];
    if (lane == 0) redmax[wave] = m;
    __syncthreads();
    m = redmax[0];
#pragma unroll
    for (int w = 1; w < 8; ++w) m = fmaxf(m, redmax[w]);

    // ---- exp + block sum ----
    float e[4];
    float s = 0.f;
#pragma unroll
    for (int t = 0; t < 4; ++t) { e[t] = __expf(v[t] - m); s += e[t]; }
#pragma unroll
    for (int off = 16; off > 0; off >>= 1)
        s += __shfl_xor(s, off, 32);
    if (lane == 0) redsum[wave] = s;
    __syncthreads();
    float stot = 0.f;
#pragma unroll
    for (int w = 0; w < 8; ++w) stot += redsum[w];
    const float inv = 1.0f / stot;

    // ---- att = softmax + 1 ;  A = att * x  (bf16, re-read by GEMM) ----
#pragma unroll
    for (int t = 0; t < 4; ++t) {
        const int d = tid + t * 256;
        const float xv  = __builtin_nontemporal_load(xrow + d);
        const float att = e[t] * inv + 1.0f;
        Abf[(size_t)row * D_DIM + d] = (__bf16)(att * xv);
    }
}

// ---------------------------------------------------------------------------
// Kernel 2: out (B x C, f32) = A (B x D, bf16) @ W^T (W: C x D, bf16)
// Block = 256 threads = 8 waves arranged 4(M) x 2(N), block tile 256 x 256.
// Wave tile 64 x 128: 4x8 fragments of 16x16 f32 accumulators (256 VGPRs).
// Fragment layouts per CDNA5 ISA §7.12.2 (wave32).
// ---------------------------------------------------------------------------
__global__ __launch_bounds__(256, 1) void gemm_wmma_kernel(
    const __bf16* __restrict__ Abf, const __bf16* __restrict__ Wbf,
    float* __restrict__ out) {
    const int lane = threadIdx.x & 31;
    const int wave = threadIdx.x >> 5;
    const int lr   = lane & 15;   // row/col within 16
    const int lh   = lane >> 4;   // half-wave select
    const int wm   = wave & 3;    // 0..3  (M)
    const int wn   = wave >> 2;   // 0..1  (N)

    const int m0 = blockIdx.x * 256 + wm * 64;   // output row base of wave tile
    const int n0 = blockIdx.y * 256 + wn * 128;  // output col base of wave tile

    v8f acc[4][8];
#pragma unroll
    for (int i = 0; i < 4; ++i)
#pragma unroll
        for (int j = 0; j < 8; ++j) {
            v8f z = {};
            acc[i][j] = z;
        }

    // A lane base: row m0+lr, K chunk offset lh*8 (K..K+7 / K+16..K+23)
    const __bf16* aBase = Abf + (size_t)(m0 + lr) * D_DIM + lh * 8;
    // B lane base: W row (= output col) n0+lr, K chunk offset lh*8 (K..K+7 / K+16..K+23)
    const __bf16* bBase = Wbf + (size_t)(n0 + lr) * D_DIM + lh * 8;

    for (int k = 0; k < D_DIM; k += 32) {
        Frag a[4], b[8];
#pragma unroll
        for (int i = 0; i < 4; ++i) {
            const uint4* p = (const uint4*)(aBase + (size_t)(i * 16) * D_DIM + k);
            a[i].q[0] = p[0];   // K = kb   .. kb+7
            a[i].q[1] = p[2];   // K = kb+16.. kb+23
        }
#pragma unroll
        for (int j = 0; j < 8; ++j) {
            const uint4* p = (const uint4*)(bBase + (size_t)(j * 16) * D_DIM + k);
            b[j].q[0] = p[0];   // K = kb   .. kb+7
            b[j].q[1] = p[2];   // K = kb+16.. kb+23
        }
#pragma unroll
        for (int i = 0; i < 4; ++i)
#pragma unroll
            for (int j = 0; j < 8; ++j)
                acc[i][j] = __builtin_amdgcn_wmma_f32_16x16x32_bf16(
                    /*neg_a=*/false, a[i].v, /*neg_b=*/false, b[j].v,
                    /*c_mod=*/(short)0, acc[i][j],
                    /*reuse_a=*/false, /*reuse_b=*/false);
    }

    // C/D layout: VGPR v, lanes 0-15 -> (M=v, N=lane); lanes 16-31 -> (M=v+8)
    // 128MB write-once stream: non-temporal so it does not rinse L2.
#pragma unroll
    for (int i = 0; i < 4; ++i) {
        const int row = m0 + i * 16 + lh * 8;
#pragma unroll
        for (int j = 0; j < 8; ++j) {
            const int col = n0 + j * 16 + lr;
            float* o = out + (size_t)row * C_DIM + col;
#pragma unroll
            for (int v = 0; v < 8; ++v)
                __builtin_nontemporal_store(acc[i][j][v], o + (size_t)v * C_DIM);
        }
    }
}

// ---------------------------------------------------------------------------
extern "C" void kernel_launch(void* const* d_in, const int* in_sizes, int n_in,
                              void* d_out, int out_size, void* d_ws, size_t ws_size,
                              hipStream_t stream) {
    const float* x   = (const float*)d_in[0];
    const float* mod = (const float*)d_in[1];
    const float* W   = (const float*)d_in[2];
    float* out = (float*)d_out;

    __bf16* Abf = (__bf16*)d_ws;                                        // 16 MB
    __bf16* Wbf = (__bf16*)((char*)d_ws + (size_t)B_ROWS * D_DIM * 2);  //  8 MB

    // W -> bf16 : 4M elements, 4/thread
    wcvt_kernel<<<(C_DIM * D_DIM) / 1024, 256, 0, stream>>>(W, Wbf);
    // fused softmax+1 scale -> bf16 A
    softmax_scale_kernel<<<B_ROWS, 256, 0, stream>>>(x, mod, Abf);
    // WMMA GEMM, block tile 256x256
    gemm_wmma_kernel<<<dim3(B_ROWS / 256, C_DIM / 256), 256, 0, stream>>>(Abf, Wbf, out);
}
